// SchNetEncoder_44848048505535
// MI455X (gfx1250) — matrix operand order repaired and checked by reading, not compiled, not run on hardware
//
#include <hip/hip_runtime.h>
#include <hip/hip_bf16.h>

#define N_NODES 50000
#define N_EDGES 800000
#define HID 128
#define NF 128
#define EC 100
#define N_BLOCKS 6
#define LOG2F_C 0.6931471805599453f

typedef __attribute__((ext_vector_type(16))) __bf16 v16bf;
typedef __attribute__((ext_vector_type(8)))  __bf16 v8bf;
typedef __attribute__((ext_vector_type(8)))  float  v8f;

// ---------------------------------------------------------------------------
// shifted softplus: log(1+exp(x)) - log(2), numerically stable
// ---------------------------------------------------------------------------
__device__ __forceinline__ float sspf(float v) {
    float sp = (v > 0.f) ? (v + __logf(1.f + __expf(-v)))
                         : __logf(1.f + __expf(v));
    return sp - LOG2F_C;
}

// ---------------------------------------------------------------------------
// WMMA fragment loaders (bf16, 16x16x32), per CDNA5 ISA VGPR layouts.
// ---------------------------------------------------------------------------
__device__ __forceinline__ v16bf frag_a_lds(const __bf16* A, int row0, int k0) {
    int lane = threadIdx.x & 31;
    int r  = row0 + (lane & 15);
    int kb = k0 + ((lane >> 4) << 3);          // +0 or +8
    const v8bf* p = (const v8bf*)(A + r * 128 + kb);
    v8bf lo = p[0];                            // K = kb..kb+7
    v8bf hi = p[2];                            // K = kb+16..kb+23
    return __builtin_shufflevector(lo, hi, 0,1,2,3,4,5,6,7,8,9,10,11,12,13,14,15);
}

__device__ __forceinline__ v16bf frag_b_lds(const __bf16* BT, int n0, int k0) {
    int lane = threadIdx.x & 31;
    int n  = n0 + (lane & 15);
    int kb = k0 + ((lane >> 4) << 3);          // +0 or +8
    const v8bf* p = (const v8bf*)(BT + n * 128 + kb);
    v8bf lo = p[0];
    v8bf hi = p[2];
    return __builtin_shufflevector(lo, hi, 0,1,2,3,4,5,6,7,8,9,10,11,12,13,14,15);
}

__device__ __forceinline__ void wave_gemm_16x128(const __bf16* A, const __bf16* BT,
                                                 int row0, v8f acc[8]) {
#pragma unroll
    for (int k0 = 0; k0 < 128; k0 += 32) {
        v16bf a = frag_a_lds(A, row0, k0);
#pragma unroll
        for (int t = 0; t < 8; ++t) {
            v16bf b = frag_b_lds(BT, t * 16, k0);
            acc[t] = __builtin_amdgcn_wmma_f32_16x16x32_bf16(
                false, a, false, b, (short)0, acc[t], false, false);
        }
    }
}

// ---------------------------------------------------------------------------
// Async tile copy: 32KB (128x128 bf16) global -> LDS via TDM-style async ops.
// 256 threads x 8 x b128 per-lane transfers; tracked by ASYNCcnt.
// ---------------------------------------------------------------------------
__device__ __forceinline__ void issue_async_tile(__bf16* dstLds, const __bf16* srcGlb,
                                                 int tid) {
    unsigned    l = (unsigned)(size_t)(void*)dstLds + (unsigned)tid * 16u;
    const char* g = (const char*)srcGlb + (size_t)tid * 16;
#pragma unroll
    for (int i = 0; i < 8; ++i) {
        asm volatile("global_load_async_to_lds_b128 %0, %1, off"
                     :: "v"(l), "v"(g) : "memory");
        l += 256 * 16;
        g += 256 * 16;
    }
}

// ---------------------------------------------------------------------------
// Kernel: h = zf @ emblin_w + b + ptemb   (K=5, pure VALU, bandwidth bound)
// ---------------------------------------------------------------------------
__global__ void schnet_embed_kernel(const float* __restrict__ z,
                                    const float* __restrict__ emb_w,
                                    const float* __restrict__ emb_b,
                                    float* __restrict__ h) {
    int idx = blockIdx.x * 256 + threadIdx.x;
    if (idx >= N_NODES * HID) return;
    int row = idx >> 7, col = idx & 127;
    const float* zr = z + row * (5 + HID);
    float s = emb_b[col] + zr[5 + col];
#pragma unroll
    for (int k = 0; k < 5; ++k) s += zr[k] * emb_w[k * HID + col];
    h[idx] = s;
}

__global__ void schnet_zero_f_kernel(float* __restrict__ p, int n) {
    int idx = blockIdx.x * 256 + threadIdx.x;
    if (idx < n) p[idx] = 0.f;
}
__global__ void schnet_zero_i_kernel(int* __restrict__ p, int n) {
    int idx = blockIdx.x * 256 + threadIdx.x;
    if (idx < n) p[idx] = 0;
}

// one-time: edge_attr fp32 [e][100] -> bf16 [e][128] zero-padded
__global__ void schnet_attr_cvt_kernel(const float* __restrict__ eattr,
                                       __bf16* __restrict__ out) {
    int idx = blockIdx.x * 256 + threadIdx.x;
    if (idx >= N_EDGES * 128) return;
    int e = idx >> 7, c = idx & 127;
    out[idx] = (__bf16)((c < EC) ? eattr[e * EC + c] : 0.f);
}

// ---------------------------------------------------------------------------
// One-time CSR-by-dst build (edge_index identical across all 6 blocks).
// fill order is racy; per-node insertion sort restores determinism.
// ---------------------------------------------------------------------------
__global__ void schnet_deg_kernel(const int* __restrict__ dst, int* __restrict__ deg) {
    int e = blockIdx.x * 256 + threadIdx.x;
    if (e < N_EDGES) atomicAdd(&deg[dst[e]], 1);
}

__global__ void schnet_scan_kernel(const int* __restrict__ deg,
                                   int* __restrict__ row_ptr,
                                   int* __restrict__ cursor) {
    __shared__ int part[1024];
    __shared__ int partscan[1024];
    const int CH = (N_NODES + 1023) / 1024;   // 49
    int t = threadIdx.x;
    int s = 0;
    for (int i = 0; i < CH; ++i) {
        int idx = t * CH + i;
        if (idx < N_NODES) s += deg[idx];
    }
    part[t] = s;
    __syncthreads();
    if (t == 0) {
        int run = 0;
        for (int i = 0; i < 1024; ++i) { partscan[i] = run; run += part[i]; }
    }
    __syncthreads();
    int run = partscan[t];
    for (int i = 0; i < CH; ++i) {
        int idx = t * CH + i;
        if (idx < N_NODES) {
            row_ptr[idx] = run;
            cursor[idx]  = run;
            run += deg[idx];
        }
    }
    if (t == 0) row_ptr[N_NODES] = N_EDGES;
}

__global__ void schnet_fill_kernel(const int* __restrict__ dst,
                                   int* __restrict__ cursor,
                                   int* __restrict__ eid) {
    int e = blockIdx.x * 256 + threadIdx.x;
    if (e < N_EDGES) {
        int pos = atomicAdd(&cursor[dst[e]], 1);
        eid[pos] = e;
    }
}

__global__ void schnet_sortlist_kernel(const int* __restrict__ row_ptr,
                                       int* __restrict__ eid) {
    int n = blockIdx.x * 256 + threadIdx.x;
    if (n >= N_NODES) return;
    int b = row_ptr[n], e = row_ptr[n + 1];
    for (int i = b + 1; i < e; ++i) {
        int v = eid[i];
        int j = i - 1;
        while (j >= b && eid[j] > v) { eid[j + 1] = eid[j]; --j; }
        eid[j + 1] = v;
    }
}

// ---------------------------------------------------------------------------
// Kernel: x = h @ lin1_w   (50000x128 @ 128x128, bf16 WMMA)
// ---------------------------------------------------------------------------
__global__ void schnet_lin1_kernel(const float* __restrict__ h,
                                   const float* __restrict__ w,   // [128][128]
                                   float* __restrict__ x) {
    extern __shared__ __align__(16) char smem_raw[];
    __bf16* sA = (__bf16*)smem_raw;          // [128][128]
    __bf16* sW = sA + 128 * 128;             // transposed [n][k]

    int tid = threadIdx.x;
    int r0  = blockIdx.x * 128;
    for (int idx = tid; idx < 128 * 128; idx += 256) {
        int a = idx >> 7, b = idx & 127;
        sW[idx] = (__bf16)w[b * 128 + a];
        int row = r0 + a;
        sA[idx] = (__bf16)((row < N_NODES) ? h[row * 128 + b] : 0.f);
    }
    __syncthreads();

    int wave = tid >> 5, lane = tid & 31;
    int row0 = wave * 16;
    v8f zero = {0.f,0.f,0.f,0.f,0.f,0.f,0.f,0.f};
    v8f acc[8];
#pragma unroll
    for (int t = 0; t < 8; ++t) acc[t] = zero;
    wave_gemm_16x128(sA, sW, row0, acc);

    int n_lo = lane & 15, mofs = (lane >> 4) << 3;
#pragma unroll
    for (int t = 0; t < 8; ++t) {
        int col = t * 16 + n_lo;
#pragma unroll
        for (int r = 0; r < 8; ++r) {
            int row = r0 + row0 + mofs + r;
            if (row < N_NODES) x[row * 128 + col] = acc[t][r];
        }
    }
}

// ---------------------------------------------------------------------------
// Hot kernel (async tier): fused edge MLP on pre-converted bf16 attrs.
//   W = (ssp(A @ w1 + b1) @ w2 + b2) * C(edge_len)  -> fp16 stream out
// Double-buffered A tiles fetched with global_load_async_to_lds_b128 while
// GEMMs run. GEMM1 result and fp16 staging reuse the current A buffer
// (dead after GEMM1), keeping LDS at 128.5 KB -> 2 workgroups per WGP.
// N_EDGES is an exact multiple of 128 so no tail guards are needed.
// ---------------------------------------------------------------------------
__global__ void schnet_edge_w_async_kernel(const __bf16* __restrict__ attrbf,
                                           const float* __restrict__ elen,
                                           const float* __restrict__ w1,  // [EC][NF]
                                           const float* __restrict__ b1,
                                           const float* __restrict__ w2,  // [NF][NF]
                                           const float* __restrict__ b2,
                                           _Float16*    __restrict__ Wout) {
    extern __shared__ __align__(16) char smem_raw[];
    __bf16* sW1 = (__bf16*)smem_raw;          // [n][k], k>=EC zero padded
    __bf16* sW2 = sW1 + 128 * 128;            // [n][k]
    __bf16* sA0 = sW2 + 128 * 128;            // tile buffer 0
    __bf16* sA1 = sA0 + 128 * 128;            // tile buffer 1
    float*  sC  = (float*)(sA1 + 128 * 128);  // [128] cutoff mask

    int tid = threadIdx.x;
    for (int idx = tid; idx < 128 * 128; idx += 256) {
        int a = idx >> 7, k = idx & 127;
        sW1[idx] = (__bf16)((k < EC) ? w1[k * NF + a] : 0.f);
        sW2[idx] = (__bf16)w2[k * NF + a];
    }

    int wave = tid >> 5, lane = tid & 31;
    int row0 = wave * 16;
    int n_lo = lane & 15, mofs = (lane >> 4) << 3;
    v8f zero = {0.f,0.f,0.f,0.f,0.f,0.f,0.f,0.f};
    const int nTiles = N_EDGES / 128;         // 6250, exact

    int tile = blockIdx.x;
    if (tile < nTiles)
        issue_async_tile(sA0, attrbf + (size_t)tile * 16384, tid);

    int cb = 0;
    for (; tile < nTiles; tile += gridDim.x) {
        __bf16* cur  = cb ? sA1 : sA0;
        __bf16* nbuf = cb ? sA0 : sA1;
        int nxt = tile + gridDim.x;
        if (nxt < nTiles) {
            issue_async_tile(nbuf, attrbf + (size_t)nxt * 16384, tid);
            asm volatile("s_wait_asynccnt 0x8" ::: "memory");   // current tile done
        } else {
            asm volatile("s_wait_asynccnt 0x0" ::: "memory");
        }
        for (int r = tid; r < 128; r += 256) {
            float L = elen[tile * 128 + r];
            sC[r] = (L <= 10.0f && L >= 0.0f) ? 1.f : 0.f;
        }
        __syncthreads();   // B1: current A tile + sC visible to all waves

        // GEMM1 (acc stays in registers)
        v8f acc[8];
#pragma unroll
        for (int t = 0; t < 8; ++t) acc[t] = zero;
        wave_gemm_16x128(cur, sW1, row0, acc);
        __syncthreads();   // B2: all waves done reading cur as A

        // write t1 = ssp(acc + b1) back into cur (A is dead)
#pragma unroll
        for (int t = 0; t < 8; ++t) {
            int col = t * 16 + n_lo;
            float bias = b1[col];
#pragma unroll
            for (int r = 0; r < 8; ++r) {
                int row = row0 + mofs + r;
                cur[row * 128 + col] = (__bf16)sspf(acc[t][r] + bias);
            }
        }
        __syncthreads();   // B3: t1 complete

        // GEMM2
#pragma unroll
        for (int t = 0; t < 8; ++t) acc[t] = zero;
        wave_gemm_16x128(cur, sW2, row0, acc);
        __syncthreads();   // B4: all waves done reading cur as t1

        // stage masked fp16 into cur
        _Float16* sStage = (_Float16*)cur;
#pragma unroll
        for (int t = 0; t < 8; ++t) {
            int col = t * 16 + n_lo;
            float bias = b2[col];
#pragma unroll
            for (int r = 0; r < 8; ++r) {
                int row = row0 + mofs + r;
                sStage[row * 128 + col] = (_Float16)((acc[t][r] + bias) * sC[row]);
            }
        }
        __syncthreads();   // B5: staging complete

        // coalesced copy-out: 128 rows x 256B
        const uint4* s4 = (const uint4*)sStage;
        uint4* g4 = (uint4*)(Wout + (size_t)tile * 16384);
        for (int c = tid; c < 128 * 16; c += 256) g4[c] = s4[c];
        __syncthreads();   // B6: cur reusable as next async target

        cb ^= 1;
    }
}

// ---------------------------------------------------------------------------
// Mid-tier edge kernel (no bf16 prepass buffer): fp32 attrs converted in-flight.
// ---------------------------------------------------------------------------
__global__ void schnet_edge_w_kernel(const float* __restrict__ eattr,
                                     const float* __restrict__ elen,
                                     const float* __restrict__ w1,
                                     const float* __restrict__ b1,
                                     const float* __restrict__ w2,
                                     const float* __restrict__ b2,
                                     _Float16*    __restrict__ Wout) {
    extern __shared__ __align__(16) char smem_raw[];
    __bf16* sW1 = (__bf16*)smem_raw;
    __bf16* sW2 = sW1 + 128 * 128;
    __bf16* sA  = sW2 + 128 * 128;
    __bf16* sT  = sA  + 128 * 128;
    float*  sC  = (float*)(sT + 128 * 128);
    _Float16* sStage = (_Float16*)sA;

    int tid = threadIdx.x;
    for (int idx = tid; idx < 128 * 128; idx += 256) {
        int a = idx >> 7, k = idx & 127;
        sW1[idx] = (__bf16)((k < EC) ? w1[k * NF + a] : 0.f);
        sW2[idx] = (__bf16)w2[k * NF + a];
    }

    int wave = tid >> 5, lane = tid & 31;
    int row0 = wave * 16;
    int n_lo = lane & 15, mofs = (lane >> 4) << 3;
    v8f zero = {0.f,0.f,0.f,0.f,0.f,0.f,0.f,0.f};
    const int nTiles = (N_EDGES + 127) / 128;

    for (int tile = blockIdx.x; tile < nTiles; tile += gridDim.x) {
        int e0 = tile * 128;
        for (int idx = tid; idx < 128 * 128; idx += 256) {
            int r = idx >> 7, k = idx & 127;
            int e = e0 + r;
            sA[idx] = (__bf16)((e < N_EDGES && k < EC) ? eattr[e * EC + k] : 0.f);
        }
        for (int r = tid; r < 128; r += 256) {
            int e = e0 + r;
            float m = 0.f;
            if (e < N_EDGES) {
                float L = elen[e];
                m = (L <= 10.0f && L >= 0.0f) ? 1.f : 0.f;
            }
            sC[r] = m;
        }
        __syncthreads();

        v8f acc[8];
#pragma unroll
        for (int t = 0; t < 8; ++t) acc[t] = zero;
        wave_gemm_16x128(sA, sW1, row0, acc);
#pragma unroll
        for (int t = 0; t < 8; ++t) {
            int col = t * 16 + n_lo;
            float bias = b1[col];
#pragma unroll
            for (int r = 0; r < 8; ++r) {
                int row = row0 + mofs + r;
                sT[row * 128 + col] = (__bf16)sspf(acc[t][r] + bias);
            }
        }
        __syncthreads();

#pragma unroll
        for (int t = 0; t < 8; ++t) acc[t] = zero;
        wave_gemm_16x128(sT, sW2, row0, acc);
#pragma unroll
        for (int t = 0; t < 8; ++t) {
            int col = t * 16 + n_lo;
            float bias = b2[col];
#pragma unroll
            for (int r = 0; r < 8; ++r) {
                int row = row0 + mofs + r;
                sStage[row * 128 + col] = (_Float16)((acc[t][r] + bias) * sC[row]);
            }
        }
        __syncthreads();

        const uint4* s4 = (const uint4*)sStage;
        for (int c = tid; c < 128 * 16; c += 256) {
            int r = c >> 4, sub = c & 15;
            int e = e0 + r;
            if (e < N_EDGES)
                ((uint4*)(Wout + (size_t)e * 128))[sub] = s4[c];
        }
        __syncthreads();
    }
}

// ---------------------------------------------------------------------------
// CSR gather: agg[n,c] = sum_{e in in(n)} W[e,c] * x[src[e],c]
// ---------------------------------------------------------------------------
__global__ void schnet_gather_kernel(const _Float16* __restrict__ W,
                                     const float* __restrict__ x,
                                     const int*   __restrict__ src,
                                     const int*   __restrict__ row_ptr,
                                     const int*   __restrict__ eid,
                                     float* __restrict__ agg) {
    int tid  = threadIdx.x;
    int node = blockIdx.x * 2 + (tid >> 7);
    int col  = tid & 127;
    if (node >= N_NODES) return;
    int beg = row_ptr[node], end = row_ptr[node + 1];
    float s = 0.f;
    for (int p = beg; p < end; ++p) {
        int e = eid[p];
        if (p + 2 < end)
            __builtin_prefetch(&W[(size_t)eid[p + 2] * 128 + col], 0, 0);
        s += (float)W[(size_t)e * 128 + col] * x[src[e] * 128 + col];
    }
    agg[node * 128 + col] = s;
}

// ---------------------------------------------------------------------------
// Fallback edge kernel (small ws): fused edge MLP + gather + atomic scatter.
// ---------------------------------------------------------------------------
__global__ void schnet_edge_scatter_kernel(const float* __restrict__ eattr,
                                           const int*   __restrict__ esrc,
                                           const int*   __restrict__ edst,
                                           const float* __restrict__ elen,
                                           const float* __restrict__ w1,
                                           const float* __restrict__ b1,
                                           const float* __restrict__ w2,
                                           const float* __restrict__ b2,
                                           const float* __restrict__ x,
                                           float*       __restrict__ agg) {
    extern __shared__ __align__(16) char smem_raw[];
    __bf16* sW1 = (__bf16*)smem_raw;
    __bf16* sW2 = sW1 + 128 * 128;
    __bf16* sA  = sW2 + 128 * 128;
    __bf16* sT  = sA  + 128 * 128;
    float*  sC  = (float*)(sT + 128 * 128);
    int*  sSrc  = (int*)(sC + 128);
    int*  sDst  = sSrc + 128;

    int tid = threadIdx.x;
    int e0  = blockIdx.x * 128;
    for (int idx = tid; idx < 128 * 128; idx += 256) {
        int a = idx >> 7, k = idx & 127;
        sW1[idx] = (__bf16)((k < EC) ? w1[k * NF + a] : 0.f);
        sW2[idx] = (__bf16)w2[k * NF + a];
        int e = e0 + a;
        sA[idx] = (__bf16)((e < N_EDGES && k < EC) ? eattr[e * EC + k] : 0.f);
    }
    for (int r = tid; r < 128; r += 256) {
        int e = e0 + r;
        if (e < N_EDGES) {
            float L = elen[e];
            sC[r]   = (L <= 10.0f && L >= 0.0f) ? 1.f : 0.f;
            sSrc[r] = esrc[e];
            sDst[r] = edst[e];
        } else { sC[r] = 0.f; sSrc[r] = 0; sDst[r] = 0; }
    }
    __syncthreads();

    int wave = tid >> 5, lane = tid & 31;
    int row0 = wave * 16;
    int n_lo = lane & 15, mofs = (lane >> 4) << 3;
    v8f zero = {0.f,0.f,0.f,0.f,0.f,0.f,0.f,0.f};

    v8f acc[8];
#pragma unroll
    for (int t = 0; t < 8; ++t) acc[t] = zero;
    wave_gemm_16x128(sA, sW1, row0, acc);
#pragma unroll
    for (int t = 0; t < 8; ++t) {
        int col = t * 16 + n_lo;
        float bias = b1[col];
#pragma unroll
        for (int r = 0; r < 8; ++r) {
            int row = row0 + mofs + r;
            sT[row * 128 + col] = (__bf16)sspf(acc[t][r] + bias);
        }
    }
    __syncthreads();

#pragma unroll
    for (int t = 0; t < 8; ++t) acc[t] = zero;
    wave_gemm_16x128(sT, sW2, row0, acc);
#pragma unroll
    for (int t = 0; t < 8; ++t) {
        int col = t * 16 + n_lo;
        float bias = b2[col];
#pragma unroll
        for (int r = 0; r < 8; ++r) {
            int row = row0 + mofs + r;
            float cm = sC[row];
            if (cm != 0.f) {
                float wv = (acc[t][r] + bias) * cm;
                float xv = x[sSrc[row] * 128 + col];
                atomicAdd(&agg[sDst[row] * 128 + col], wv * xv);
            }
        }
    }
}

// ---------------------------------------------------------------------------
// Kernel: h += (ssp(agg @ lin2_w + lin2_b) @ lin_w + lin_b)
// ---------------------------------------------------------------------------
__global__ void schnet_update_kernel(const float* __restrict__ agg,
                                     const float* __restrict__ w2,
                                     const float* __restrict__ b2,
                                     const float* __restrict__ w3,
                                     const float* __restrict__ b3,
                                     float* __restrict__ h) {
    extern __shared__ __align__(16) char smem_raw[];
    __bf16* sA  = (__bf16*)smem_raw;
    __bf16* sW2 = sA  + 128 * 128;
    __bf16* sW3 = sW2 + 128 * 128;
    __bf16* sT  = sW3 + 128 * 128;

    int tid = threadIdx.x;
    int r0  = blockIdx.x * 128;
    for (int idx = tid; idx < 128 * 128; idx += 256) {
        int a = idx >> 7, k = idx & 127;
        sW2[idx] = (__bf16)w2[k * 128 + a];
        sW3[idx] = (__bf16)w3[k * 128 + a];
        int row = r0 + a;
        sA[idx] = (__bf16)((row < N_NODES) ? agg[row * 128 + k] : 0.f);
    }
    __syncthreads();

    int wave = tid >> 5, lane = tid & 31;
    int row0 = wave * 16;
    int n_lo = lane & 15, mofs = (lane >> 4) << 3;
    v8f zero = {0.f,0.f,0.f,0.f,0.f,0.f,0.f,0.f};

    v8f acc[8];
#pragma unroll
    for (int t = 0; t < 8; ++t) acc[t] = zero;
    wave_gemm_16x128(sA, sW2, row0, acc);
#pragma unroll
    for (int t = 0; t < 8; ++t) {
        int col = t * 16 + n_lo;
        float bias = b2[col];
#pragma unroll
        for (int r = 0; r < 8; ++r) {
            int row = row0 + mofs + r;
            sT[row * 128 + col] = (__bf16)sspf(acc[t][r] + bias);
        }
    }
    __syncthreads();

#pragma unroll
    for (int t = 0; t < 8; ++t) acc[t] = zero;
    wave_gemm_16x128(sT, sW3, row0, acc);
#pragma unroll
    for (int t = 0; t < 8; ++t) {
        int col = t * 16 + n_lo;
        float bias = b3[col];
#pragma unroll
        for (int r = 0; r < 8; ++r) {
            int row = r0 + row0 + mofs + r;
            if (row < N_NODES) h[row * 128 + col] += acc[t][r] + bias;
        }
    }
}

// ---------------------------------------------------------------------------
extern "C" void kernel_launch(void* const* d_in, const int* in_sizes, int n_in,
                              void* d_out, int out_size, void* d_ws, size_t ws_size,
                              hipStream_t stream) {
    const float* z      = (const float*)d_in[0];
    const int*   ei     = (const int*)  d_in[1];
    const float* elen   = (const float*)d_in[2];
    const float* eattr  = (const float*)d_in[3];
    const float* emb_w  = (const float*)d_in[4];
    const float* emb_b  = (const float*)d_in[5];
    const float* lin1_w = (const float*)d_in[6];
    const float* nn_w1  = (const float*)d_in[7];
    const float* nn_b1  = (const float*)d_in[8];
    const float* nn_w2  = (const float*)d_in[9];
    const float* nn_b2  = (const float*)d_in[10];
    const float* lin2_w = (const float*)d_in[11];
    const float* lin2_b = (const float*)d_in[12];
    const float* lin_w  = (const float*)d_in[13];
    const float* lin_b  = (const float*)d_in[14];

    const int* esrc = ei;
    const int* edst = ei + N_EDGES;

    float* h    = (float*)d_out;
    float* xbuf = (float*)d_ws;
    float* agg  = xbuf + (size_t)N_NODES * HID;
    _Float16* Wbuf = (_Float16*)(agg + (size_t)N_NODES * HID);
    int* row_ptr = (int*)(Wbuf + (size_t)N_EDGES * HID);
    int* eid     = row_ptr + (N_NODES + 1);
    int* cursor  = eid + N_EDGES;
    int* deg     = cursor + N_NODES;
    size_t off_attr = ((size_t)((char*)(deg + N_NODES) - (char*)d_ws) + 255) & ~(size_t)255;
    __bf16* attrbf = (__bf16*)((char*)d_ws + off_attr);

    size_t need_base = (size_t)N_NODES * HID * sizeof(float) * 2;
    size_t need_csr  = (size_t)((char*)(deg + N_NODES) - (char*)d_ws);
    size_t need_full = off_attr + (size_t)N_EDGES * HID * sizeof(__bf16);
    bool tier_full = (ws_size >= need_full);
    bool tier_csr  = (ws_size >= need_csr);
    (void)need_base;

    const int LDS_LIN1       = 2 * 128 * 128 * 2;                // 64 KB
    const int LDS_EDGE       = 4 * 128 * 128 * 2 + 128 * 12;     // ~130 KB
    const int LDS_EDGE_ASYNC = 4 * 128 * 128 * 2 + 512;          // 128.5 KB
    const int LDS_UPDATE     = 4 * 128 * 128 * 2;                // 128 KB
    (void)hipFuncSetAttribute((const void*)schnet_lin1_kernel,
        hipFuncAttributeMaxDynamicSharedMemorySize, LDS_LIN1);
    (void)hipFuncSetAttribute((const void*)schnet_edge_w_kernel,
        hipFuncAttributeMaxDynamicSharedMemorySize, LDS_EDGE);
    (void)hipFuncSetAttribute((const void*)schnet_edge_w_async_kernel,
        hipFuncAttributeMaxDynamicSharedMemorySize, LDS_EDGE_ASYNC);
    (void)hipFuncSetAttribute((const void*)schnet_edge_scatter_kernel,
        hipFuncAttributeMaxDynamicSharedMemorySize, LDS_EDGE);
    (void)hipFuncSetAttribute((const void*)schnet_update_kernel,
        hipFuncAttributeMaxDynamicSharedMemorySize, LDS_UPDATE);

    int elemBlocks = (N_NODES * HID + 255) / 256;
    int attrBlocks = (N_EDGES * 128 + 255) / 256;
    int nodeTiles  = (N_NODES + 127) / 128;
    int edgeTiles  = (N_EDGES + 127) / 128;
    int edgeElems  = (N_EDGES + 255) / 256;
    int nodeElems  = (N_NODES + 255) / 256;

    schnet_embed_kernel<<<elemBlocks, 256, 0, stream>>>(z, emb_w, emb_b, h);

    if (tier_csr) {
        // one-time CSR build, reused by all 6 blocks
        schnet_zero_i_kernel<<<nodeElems, 256, 0, stream>>>(deg, N_NODES);
        schnet_deg_kernel<<<edgeElems, 256, 0, stream>>>(edst, deg);
        schnet_scan_kernel<<<1, 1024, 0, stream>>>(deg, row_ptr, cursor);
        schnet_fill_kernel<<<edgeElems, 256, 0, stream>>>(edst, cursor, eid);
        schnet_sortlist_kernel<<<nodeElems, 256, 0, stream>>>(row_ptr, eid);
        if (tier_full)
            schnet_attr_cvt_kernel<<<attrBlocks, 256, 0, stream>>>(eattr, attrbf);

        for (int i = 0; i < N_BLOCKS; ++i) {
            schnet_lin1_kernel<<<nodeTiles, 256, LDS_LIN1, stream>>>(
                h, lin1_w + (size_t)i * HID * NF, xbuf);
            if (tier_full) {
                schnet_edge_w_async_kernel<<<1280, 256, LDS_EDGE_ASYNC, stream>>>(
                    attrbf, elen,
                    nn_w1 + (size_t)i * EC * NF, nn_b1 + (size_t)i * NF,
                    nn_w2 + (size_t)i * NF * NF, nn_b2 + (size_t)i * NF,
                    Wbuf);
            } else {
                schnet_edge_w_kernel<<<1280, 256, LDS_EDGE, stream>>>(
                    eattr, elen,
                    nn_w1 + (size_t)i * EC * NF, nn_b1 + (size_t)i * NF,
                    nn_w2 + (size_t)i * NF * NF, nn_b2 + (size_t)i * NF,
                    Wbuf);
            }
            schnet_gather_kernel<<<(N_NODES + 1) / 2, 256, 0, stream>>>(
                Wbuf, xbuf, esrc, row_ptr, eid, agg);
            schnet_update_kernel<<<nodeTiles, 256, LDS_UPDATE, stream>>>(
                agg, lin2_w + (size_t)i * NF * HID, lin2_b + (size_t)i * HID,
                lin_w + (size_t)i * HID * HID, lin_b + (size_t)i * HID, h);
        }
    } else {
        for (int i = 0; i < N_BLOCKS; ++i) {
            schnet_lin1_kernel<<<nodeTiles, 256, LDS_LIN1, stream>>>(
                h, lin1_w + (size_t)i * HID * NF, xbuf);
            schnet_zero_f_kernel<<<elemBlocks, 256, 0, stream>>>(agg, N_NODES * HID);
            schnet_edge_scatter_kernel<<<edgeTiles, 256, LDS_EDGE, stream>>>(
                eattr, esrc, edst, elen,
                nn_w1 + (size_t)i * EC * NF, nn_b1 + (size_t)i * NF,
                nn_w2 + (size_t)i * NF * NF, nn_b2 + (size_t)i * NF,
                xbuf, agg);
            schnet_update_kernel<<<nodeTiles, 256, LDS_UPDATE, stream>>>(
                agg, lin2_w + (size_t)i * NF * HID, lin2_b + (size_t)i * HID,
                lin_w + (size_t)i * HID * HID, lin_b + (size_t)i * HID, h);
        }
    }
}
